// EGNNVectorField_79173427135042
// MI455X (gfx1250) — matrix operand order repaired and checked by reading, not compiled, not run on hardware
//
#include <hip/hip_runtime.h>
#include <hip/hip_bf16.h>

typedef __attribute__((ext_vector_type(16))) _Float16 v16h;
typedef __attribute__((ext_vector_type(8)))  _Float16 v8h;
typedef __attribute__((ext_vector_type(8)))  float    v8f;

#define B_  4
#define T_  2560
#define Q_  2048
#define G_  512
#define NN  10240      // B_*T_
#define C_  512
#define H_  512
#define KNN 8
#define E_  81920      // NN*KNN
#define MQ  8192       // B_*Q_

#define LDSROW 72      // padded row stride (halves): 144B = 36 dwords -> bank spread

__device__ __forceinline__ float siluf(float x) { return x / (1.0f + __expf(-x)); }

__device__ __forceinline__ v16h cat16(v8h lo, v8h hi) {
  return __builtin_shufflevector(lo, hi, 0,1,2,3,4,5,6,7,8,9,10,11,12,13,14,15);
}

// Async-DMA one 16B segment global->LDS (CDNA5 async engine, tracked by ASYNCcnt)
__device__ __forceinline__ void async_b128(unsigned lds_off, const _Float16* src) {
  asm volatile("global_load_async_to_lds_b128 %0, %1, off"
               :: "v"(lds_off), "v"((unsigned long long)(uintptr_t)src)
               : "memory");
}
__device__ __forceinline__ void wait_async0() {
  asm volatile("s_wait_asynccnt 0" ::: "memory");
}

// Stage a 64row x 64half A chunk into LDS: wave w moves rows [8w,8w+8), 2 instrs/lane.
// sdst is a (runtime-selected) generic pointer into __shared__; the DS offset is
// computed per-lane so no addrspacecast lands in a static initializer.
__device__ __forceinline__ void stage_A64(_Float16* sdst, const _Float16* A, int lda,
                                          int mb, int k0, int gin, int gout,
                                          int wave, int lane) {
  int srow = wave * 8 + (lane >> 2);
  int seg  = lane & 3;
  int lr = mb + srow;
  int ar = gin ? ((lr / gin) * gout + (lr % gin)) : lr;
  const _Float16* gp = A + (size_t)ar * lda + k0 + seg * 8;
  unsigned laddr = (unsigned)(uintptr_t)(sdst + srow * LDSROW + seg * 8);
  async_b128(laddr,       gp);
  async_b128(laddr + 64u, gp + 32);
}

// ---------------------------------------------------------------------------
// init: build coords x0 [NN,3], h fp32 [NN,512] (queries=0, grid=codes), h16
// ---------------------------------------------------------------------------
__global__ void init_kernel(const float* __restrict__ qp, const float* __restrict__ codes,
                            float* __restrict__ x0, float* __restrict__ h,
                            _Float16* __restrict__ h16) {
  int idx = blockIdx.x * blockDim.x + threadIdx.x;   // over NN*C_
  if (idx >= NN * C_) return;
  int node = idx >> 9, c = idx & 511;
  int b = node / T_, t = node - b * T_;
  float hv = 0.0f;
  if (t >= Q_) hv = codes[((size_t)(b * G_ + (t - Q_)) << 9) + c];
  h[idx] = hv; h16[idx] = (_Float16)hv;
  if (c < 3) {
    float xv;
    if (t < Q_) {
      xv = qp[(size_t)(b * Q_ + t) * 3 + c];
    } else {
      int g = t - Q_;
      int ii = (c == 0) ? (g >> 6) : (c == 1 ? ((g >> 3) & 7) : (g & 7));
      xv = -1.0f + (2.0f / 7.0f) * (float)ii;
    }
    x0[node * 3 + c] = xv;
  }
}

// ---------------------------------------------------------------------------
// knn: per-batch 8-NN, coords staged in LDS; sorted ascending by distance
// ---------------------------------------------------------------------------
__global__ void knn_kernel(const float* __restrict__ x0, int* __restrict__ nbr) {
  __shared__ float sx[T_ * 3];                       // 30 KB
  int b = blockIdx.y;
  for (int idx = threadIdx.x; idx < T_ * 3; idx += blockDim.x)
    sx[idx] = x0[(size_t)b * T_ * 3 + idx];
  __syncthreads();
  int i = blockIdx.x * blockDim.x + threadIdx.x;
  if (i >= T_) return;
  float xi = sx[i*3], yi = sx[i*3+1], zi = sx[i*3+2];
  float bd[KNN]; int bj[KNN];
#pragma unroll
  for (int k = 0; k < KNN; k++) { bd[k] = 3.0e38f; bj[k] = 0; }
  for (int j = 0; j < T_; j++) {
    if (j == i) continue;
    float dx = sx[j*3]-xi, dy = sx[j*3+1]-yi, dz = sx[j*3+2]-zi;
    float d2 = dx*dx + dy*dy + dz*dz;
    if (d2 < bd[KNN-1]) {
      int p = KNN - 1;
      while (p > 0 && bd[p-1] > d2) { bd[p] = bd[p-1]; bj[p] = bj[p-1]; p--; }
      bd[p] = d2; bj[p] = j;
    }
  }
  for (int k = 0; k < KNN; k++) nbr[(b*T_ + i)*KNN + k] = b*T_ + bj[k];
}

// ---------------------------------------------------------------------------
// weight prep: fp32 [K,N] (row stride srcStride) -> f16 transposed [N,K]
// ---------------------------------------------------------------------------
__global__ void transpose_to_f16(const float* __restrict__ src, _Float16* __restrict__ dst,
                                 int K, int N, int srcStride) {
  int idx = blockIdx.x * blockDim.x + threadIdx.x;
  if (idx >= K * N) return;
  int n = idx / K, k = idx - n * K;
  dst[idx] = (_Float16)src[(size_t)k * srcStride + n];
}

__global__ void copy_f32(const float* __restrict__ src, float* __restrict__ dst, int n) {
  int i = blockIdx.x * blockDim.x + threadIdx.x;
  if (i < n) dst[i] = src[i];
}

__global__ void init_coef(float* __restrict__ coef, const float* __restrict__ cb) {
  int e = blockIdx.x * blockDim.x + threadIdx.x;
  if (e < E_) coef[e] = cb[0];
}

// ---------------------------------------------------------------------------
// Generic WMMA GEMM: C[M,N] = act(A[M,K](f16) @ Bt[N,K]^T(f16) + bias + resid)
// grid = (N/128, M/64), block = 256 (8 waves).
// A strip (64 x K) async-DMA'd into double-buffered LDS (chunk Kc=64) by the
// CDNA5 async engine; fragments read back via ds_load_b128. Each wave computes
// a 64x16 strip = 4 M-subtiles sharing ONE register B fragment per K-step.
// gin/gout: optional A-row remap (query-row gather): ar=(r/gin)*gout + r%gin
// ---------------------------------------------------------------------------
__global__ void gemm_wmma(const _Float16* __restrict__ A, int lda,
                          const _Float16* __restrict__ Bt, int ldb,
                          const float* __restrict__ bias,
                          const float* __restrict__ resid,
                          float* __restrict__ outF, _Float16* __restrict__ outH,
                          int M, int N, int K, int act, int gin, int gout) {
  __shared__ _Float16 sA[2][64 * LDSROW];
  int wave = threadIdx.x >> 5;
  int lane = threadIdx.x & 31;
  int nb = blockIdx.x * 128 + wave * 16;
  int mb = blockIdx.y * 64;
  int kA = (lane >> 4) * 8;                                // A half-lane K base
  int kB = (lane >> 4) * 8;                                // B half-lane K base
  const _Float16* bp = Bt + (size_t)(nb + (lane & 15)) * ldb + kB;
  _Float16* s0 = &sA[0][0];
  _Float16* s1 = &sA[1][0];

  v8f acc[4] = {};
  stage_A64(s0, A, lda, mb, 0, gin, gout, wave, lane);
  wait_async0();
  __syncthreads();
  int nch = K >> 6;
  for (int c = 0; c < nch; c++) {
    int k0 = c << 6;
    if (c + 1 < nch)
      stage_A64(((c + 1) & 1) ? s1 : s0, A, lda, mb, k0 + 64, gin, gout, wave, lane);
    const _Float16* sc = (c & 1) ? s1 : s0;
    __builtin_prefetch((const void*)(bp + k0 + 128), 0, 1);
#pragma unroll
    for (int ks = 0; ks < 64; ks += 32) {
      v16h b = cat16(*(const v8h*)(bp + k0 + ks), *(const v8h*)(bp + k0 + ks + 16));
#pragma unroll
      for (int t = 0; t < 4; t++) {
        const _Float16* sap = sc + (t * 16 + (lane & 15)) * LDSROW + ks + kA;
        v16h a = cat16(*(const v8h*)sap, *(const v8h*)(sap + 16));
        acc[t] = __builtin_amdgcn_wmma_f32_16x16x32_f16(false, a, false, b,
                                                        (short)0, acc[t], false, false);
      }
    }
    wait_async0();        // next chunk landed (overlapped with the WMMAs above)
    __syncthreads();      // publish LDS + protect buffer reuse
  }

  int col = nb + (lane & 15);
  float bv = bias ? bias[col] : 0.0f;
#pragma unroll
  for (int t = 0; t < 4; t++) {
    int rbase = mb + t * 16 + (lane >> 4) * 8;
#pragma unroll
    for (int v = 0; v < 8; v++) {
      int row = rbase + v;
      float val = acc[t][v] + bv;
      if (resid) val += resid[(size_t)row * N + col];
      if (act == 1) val = siluf(val);
      else if (act == 2) val = fmaxf(val, 0.0f);
      if (outF) outF[(size_t)row * N + col] = val;
      if (outH) outH[(size_t)row * N + col] = (_Float16)val;
    }
  }
}

// ---------------------------------------------------------------------------
// edge m1: m1[i*8+k] = silu(hA[nbr] + hB[i] + dist*wd)  (b1 folded into hB)
// one block per node; hB row + wd staged in LDS
// ---------------------------------------------------------------------------
__global__ void edge_m1_kernel(const float* __restrict__ x, const int* __restrict__ nbr,
                               const float* __restrict__ hA, const float* __restrict__ hB,
                               const float* __restrict__ wd, _Float16* __restrict__ m1) {
  __shared__ float sB[C_], sW[C_], sD[KNN];
  __shared__ int sJ[KNN];
  int i = blockIdx.x;
  int t = threadIdx.x;
  sB[t]       = hB[(size_t)i * C_ + t];
  sB[t + 256] = hB[(size_t)i * C_ + t + 256];
  sW[t]       = wd[t];
  sW[t + 256] = wd[t + 256];
  if (t < KNN) {
    int j = nbr[i * KNN + t];
    sJ[t] = j;
    float dx = x[j*3]-x[i*3], dy = x[j*3+1]-x[i*3+1], dz = x[j*3+2]-x[i*3+2];
    sD[t] = sqrtf(dx*dx + dy*dy + dz*dz);
  }
  __syncthreads();
  for (int k = 0; k < KNN; k++) {
    int j = sJ[k];
    float d = sD[k];
    size_t ebase = ((size_t)i * KNN + k) * C_;
#pragma unroll
    for (int c = t; c < C_; c += 256) {
      float val = hA[(size_t)j * C_ + c] + sB[c] + d * sW[c];
      m1[ebase + c] = (_Float16)siluf(val);
    }
  }
}

// ---------------------------------------------------------------------------
// fused edge GEMM2: m2 = silu(m1 @ ew2 + b2).
// Same LDS-staged mainloop; 64-row M-strip = 8 nodes. Epilogue: lane-local
// mean over each node's 8 edges -> maggr (f16); coef = m2@cw via 16-lane
// shfl_xor reduce + atomicAdd. m2 never materialized.
// grid = (512/128, E/64), block 256.
// ---------------------------------------------------------------------------
__global__ void edge_gemm2_kernel(const _Float16* __restrict__ m1,
                                  const _Float16* __restrict__ Bt,   // ew2_t [512,512]
                                  const float* __restrict__ bias,    // eb2
                                  const float* __restrict__ cw,      // coord_w col
                                  _Float16* __restrict__ maggr,
                                  float* __restrict__ coef) {
  __shared__ _Float16 sA[2][64 * LDSROW];
  int wave = threadIdx.x >> 5, lane = threadIdx.x & 31;
  int nb = blockIdx.x * 128 + wave * 16;
  int mb = blockIdx.y * 64;
  int kA = (lane >> 4) * 8, kB = (lane >> 4) * 8;
  const _Float16* bp = Bt + (size_t)(nb + (lane & 15)) * H_ + kB;
  _Float16* s0 = &sA[0][0];
  _Float16* s1 = &sA[1][0];

  v8f acc[4] = {};
  stage_A64(s0, m1, H_, mb, 0, 0, 0, wave, lane);
  wait_async0();
  __syncthreads();
  const int nch = H_ >> 6;
  for (int c = 0; c < nch; c++) {
    int k0 = c << 6;
    if (c + 1 < nch)
      stage_A64(((c + 1) & 1) ? s1 : s0, m1, H_, mb, k0 + 64, 0, 0, wave, lane);
    const _Float16* sc = (c & 1) ? s1 : s0;
    __builtin_prefetch((const void*)(bp + k0 + 128), 0, 1);
#pragma unroll
    for (int ks = 0; ks < 64; ks += 32) {
      v16h b = cat16(*(const v8h*)(bp + k0 + ks), *(const v8h*)(bp + k0 + ks + 16));
#pragma unroll
      for (int t = 0; t < 4; t++) {
        const _Float16* sap = sc + (t * 16 + (lane & 15)) * LDSROW + ks + kA;
        v16h a = cat16(*(const v8h*)sap, *(const v8h*)(sap + 16));
        acc[t] = __builtin_amdgcn_wmma_f32_16x16x32_f16(false, a, false, b,
                                                        (short)0, acc[t], false, false);
      }
    }
    wait_async0();
    __syncthreads();
  }

  int col = nb + (lane & 15);
  float bv = bias[col], cwv = cw[col];
#pragma unroll
  for (int t = 0; t < 4; t++) {
    float p[8], sum = 0.0f;
#pragma unroll
    for (int v = 0; v < 8; v++) {
      float val = siluf(acc[t][v] + bv);            // m2[edge v of my node][col]
      sum += val;
      p[v] = val * cwv;
    }
    int node = ((mb + t * 16) >> 3) + (lane >> 4);  // rows 0-7 -> node0, 8-15 -> node0+1
    maggr[(size_t)node * H_ + col] = (_Float16)(sum * 0.125f);
#pragma unroll
    for (int v = 0; v < 8; v++) {                   // reduce over 16 columns of tile
      p[v] += __shfl_xor(p[v], 1, 32);
      p[v] += __shfl_xor(p[v], 2, 32);
      p[v] += __shfl_xor(p[v], 4, 32);
      p[v] += __shfl_xor(p[v], 8, 32);
    }
    if ((lane & 15) == 0) {
      int ebase = mb + t * 16 + (lane >> 4) * 8;
#pragma unroll
      for (int v = 0; v < 8; v++) atomicAdd(&coef[ebase + v], p[v]);
    }
  }
}

// ---------------------------------------------------------------------------
// coord update: x_next = x + mean_k coef * unit(rel), using OLD x
// ---------------------------------------------------------------------------
__global__ void coord_update_kernel(const float* __restrict__ xc, const int* __restrict__ nbr,
                                    const float* __restrict__ coef, float* __restrict__ xn) {
  int i = blockIdx.x * blockDim.x + threadIdx.x;
  if (i >= NN) return;
  float xi = xc[i*3], yi = xc[i*3+1], zi = xc[i*3+2];
  float dx = 0, dy = 0, dz = 0;
#pragma unroll
  for (int k = 0; k < KNN; k++) {
    int j = nbr[i*KNN + k];
    float rx = xc[j*3]-xi, ry = xc[j*3+1]-yi, rz = xc[j*3+2]-zi;
    float d = sqrtf(rx*rx + ry*ry + rz*rz);
    float inv = 1.0f / (d + 1e-8f);
    float c = coef[i*KNN + k];
    dx += c*rx*inv; dy += c*ry*inv; dz += c*rz*inv;
  }
  xn[i*3] = xi + dx*0.125f; xn[i*3+1] = yi + dy*0.125f; xn[i*3+2] = zi + dz*0.125f;
}

// ---------------------------------------------------------------------------
// final field head: out = x_q - (f @ fw2 + fb2); one wave per query row
// ---------------------------------------------------------------------------
__global__ void field_out_kernel(const float* __restrict__ f, const float* __restrict__ w2,
                                 const float* __restrict__ b2, const float* __restrict__ xfin,
                                 float* __restrict__ out) {
  int r = (blockIdx.x * blockDim.x + threadIdx.x) >> 5;
  int lane = threadIdx.x & 31;
  if (r >= MQ) return;
  int node = (r / Q_) * T_ + (r % Q_);
  float acc[15];
#pragma unroll
  for (int a = 0; a < 15; a++) acc[a] = 0.0f;
  for (int q = 0; q < 16; q++) {
    int c = lane + (q << 5);
    float fv = f[(size_t)r * H_ + c];
#pragma unroll
    for (int a = 0; a < 15; a++) acc[a] += fv * w2[c*15 + a];
  }
#pragma unroll
  for (int a = 0; a < 15; a++) {
    acc[a] += __shfl_xor(acc[a], 1, 32);
    acc[a] += __shfl_xor(acc[a], 2, 32);
    acc[a] += __shfl_xor(acc[a], 4, 32);
    acc[a] += __shfl_xor(acc[a], 8, 32);
    acc[a] += __shfl_xor(acc[a], 16, 32);
  }
  if (lane == 0) {
#pragma unroll
    for (int a = 0; a < 15; a++) {
      int c3 = a % 3;
      out[(size_t)r * 15 + a] = xfin[node*3 + c3] - (acc[a] + b2[a]);
    }
  }
}

// ---------------------------------------------------------------------------
extern "C" void kernel_launch(void* const* d_in, const int* in_sizes, int n_in,
                              void* d_out, int out_size, void* d_ws, size_t ws_size,
                              hipStream_t stream) {
  (void)in_sizes; (void)n_in; (void)out_size; (void)ws_size;
  const float* qp    = (const float*)d_in[0];
  const float* codes = (const float*)d_in[1];
  const float* ew1   = (const float*)d_in[2];
  const float* eb1   = (const float*)d_in[3];
  const float* ew2   = (const float*)d_in[4];
  const float* eb2   = (const float*)d_in[5];
  const float* cw    = (const float*)d_in[6];
  const float* cb    = (const float*)d_in[7];
  const float* nw1   = (const float*)d_in[8];
  const float* nb1   = (const float*)d_in[9];
  const float* nw2   = (const float*)d_in[10];
  const float* nb2   = (const float*)d_in[11];
  const float* fw1   = (const float*)d_in[12];
  const float* fb1   = (const float*)d_in[13];
  const float* fw2   = (const float*)d_in[14];
  const float* fb2   = (const float*)d_in[15];
  float* out = (float*)d_out;

  size_t off = 0;
  auto alloc = [&](size_t bytes) -> void* {
    void* p = (char*)d_ws + off;
    off += (bytes + 255) & ~(size_t)255;
    return p;
  };
  float*     x0    = (float*)alloc((size_t)NN*3*4);
  float*     x1    = (float*)alloc((size_t)NN*3*4);
  float*     h     = (float*)alloc((size_t)NN*C_*4);
  _Float16*  h16   = (_Float16*)alloc((size_t)NN*C_*2);
  float*     hA    = (float*)alloc((size_t)NN*H_*4);
  float*     hB    = (float*)alloc((size_t)NN*H_*4);
  _Float16*  m1    = (_Float16*)alloc((size_t)E_*H_*2);
  _Float16*  maggr = (_Float16*)alloc((size_t)NN*H_*2);
  float*     tmpN  = (float*)alloc((size_t)NN*H_*4);
  _Float16*  hd16  = (_Float16*)alloc((size_t)NN*H_*2);
  float*     coef  = (float*)alloc((size_t)E_*4);
  int*       nbr   = (int*)alloc((size_t)NN*KNN*4);
  float*     fbuf  = (float*)alloc((size_t)MQ*H_*4);
  _Float16 *Wa_t[3], *Wb_t[3], *ew2_t[3], *nw1_t[3], *nw2_t[3];
  float* wd_t[3];
  for (int l = 0; l < 3; l++) {
    Wa_t[l]  = (_Float16*)alloc((size_t)512*512*2);
    Wb_t[l]  = (_Float16*)alloc((size_t)512*512*2);
    ew2_t[l] = (_Float16*)alloc((size_t)512*512*2);
    nw1_t[l] = (_Float16*)alloc((size_t)512*1024*2);
    nw2_t[l] = (_Float16*)alloc((size_t)512*512*2);
    wd_t[l]  = (float*)alloc((size_t)512*4);
  }
  _Float16* fw1_t = (_Float16*)alloc((size_t)512*512*2);

  // ---- weight prep (f16 transposed [N,K]) ----
  int tpg = (512*512 + 255) / 256;
  int tpg2 = (1024*512 + 255) / 256;
  for (int l = 0; l < 3; l++) {
    const float* e1l = ew1 + (size_t)l * 1025 * 512;
    transpose_to_f16<<<tpg, 256, 0, stream>>>(e1l,            Wa_t[l], 512, 512, 512);
    transpose_to_f16<<<tpg, 256, 0, stream>>>(e1l + 512*512,  Wb_t[l], 512, 512, 512);
    copy_f32<<<2, 256, 0, stream>>>(e1l + 1024*512, wd_t[l], 512);
    transpose_to_f16<<<tpg, 256, 0, stream>>>(ew2 + (size_t)l*512*512, ew2_t[l], 512, 512, 512);
    transpose_to_f16<<<tpg2, 256, 0, stream>>>(nw1 + (size_t)l*1024*512, nw1_t[l], 1024, 512, 512);
    transpose_to_f16<<<tpg, 256, 0, stream>>>(nw2 + (size_t)l*512*512, nw2_t[l], 512, 512, 512);
  }
  transpose_to_f16<<<tpg, 256, 0, stream>>>(fw1, fw1_t, 512, 512, 512);

  // ---- init + knn ----
  init_kernel<<<(NN*C_ + 255)/256, 256, 0, stream>>>(qp, codes, x0, h, h16);
  knn_kernel<<<dim3(T_/256, B_), 256, 0, stream>>>(x0, nbr);

  float* xc = x0;
  float* xn = x1;
  for (int l = 0; l < 3; l++) {
    // hA = h@Wa ; hB = h@Wb + eb1   (node-level GEMMs, NN x 512 x 512)
    gemm_wmma<<<dim3(4, NN/64), 256, 0, stream>>>(h16, 512, Wa_t[l], 512,
        nullptr, nullptr, hA, nullptr, NN, 512, 512, 0, 0, 0);
    gemm_wmma<<<dim3(4, NN/64), 256, 0, stream>>>(h16, 512, Wb_t[l], 512,
        eb1 + (size_t)l*512, nullptr, hB, nullptr, NN, 512, 512, 0, 0, 0);
    // m1 = silu(hA[src] + hB[dst] + dist*wd)
    edge_m1_kernel<<<NN, 256, 0, stream>>>(xc, nbr, hA, hB, wd_t[l], m1);
    // coef = cb; then fused m2-GEMM adds m2@cw, emits maggr
    init_coef<<<(E_ + 255)/256, 256, 0, stream>>>(coef, cb + l);
    edge_gemm2_kernel<<<dim3(4, E_/64), 256, 0, stream>>>(m1, ew2_t[l],
        eb2 + (size_t)l*512, cw + (size_t)l*512, maggr, coef);
    // x update (reads old x, writes new)
    coord_update_kernel<<<(NN + 255)/256, 256, 0, stream>>>(xc, nbr, coef, xn);
    // node MLP: hd = silu(h@nw1[:512] + maggr@nw1[512:] + nb1); h += hd@nw2 + nb2
    gemm_wmma<<<dim3(4, NN/64), 256, 0, stream>>>(h16, 512, nw1_t[l], 1024,
        nullptr, nullptr, tmpN, nullptr, NN, 512, 512, 0, 0, 0);
    gemm_wmma<<<dim3(4, NN/64), 256, 0, stream>>>(maggr, 512, nw1_t[l] + 512, 1024,
        nb1 + (size_t)l*512, tmpN, nullptr, hd16, NN, 512, 512, 1, 0, 0);
    gemm_wmma<<<dim3(4, NN/64), 256, 0, stream>>>(hd16, 512, nw2_t[l], 512,
        nb2 + (size_t)l*512, h, h, h16, NN, 512, 512, 0, 0, 0);
    float* tswap = xc; xc = xn; xn = tswap;
  }

  // ---- field head on query rows (gathered via gin/gout remap) ----
  gemm_wmma<<<dim3(4, MQ/64), 256, 0, stream>>>(h16, 512, fw1_t, 512,
      fb1, nullptr, fbuf, nullptr, MQ, 512, 512, 2, Q_, T_);
  field_out_kernel<<<MQ/8, 256, 0, stream>>>(fbuf, fw2, fb2, xc, out);
}
